// LocalFeatureExtractor_16492674416928
// MI455X (gfx1250) — matrix operand order repaired and checked by reading, not compiled, not run on hardware
//
#include <hip/hip_runtime.h>

typedef __attribute__((ext_vector_type(16))) _Float16 v16h;
typedef __attribute__((ext_vector_type(8)))  _Float16 v8h;
typedef __attribute__((ext_vector_type(4)))  _Float16 v4h;
typedef __attribute__((ext_vector_type(8)))  float    v8f;

#define C_DIM 128
#define N_DIM 20000
#define B_DIM 8
#define K_NB  10
#define SLOTS 11                 // self + 10 neighbors
#define KTOT  (SLOTS * C_DIM)    // 1408 = GEMM reduction dim

// ---------------------------------------------------------------------------
// Kernel 1: fold W1, Wc, W2 into 11 effective 128x128 matrices (f16) + bias.
//   M_0[d][c] = W1[d][c] + sum_e W2[d][e]*Wc[e][c][0]
//   M_k[d][c] =            sum_e W2[d][e]*Wc[e][c][k]   (k=1..10)
//   bias[d]   = W1_b[d] + W2_b[d] + sum_e W2[d][e]*Wc_b[e]
// ---------------------------------------------------------------------------
__global__ __launch_bounds__(128) void prep_weights(
    const float* __restrict__ W1w, const float* __restrict__ W1b,
    const float* __restrict__ Wcw, const float* __restrict__ Wcb,
    const float* __restrict__ W2w, const float* __restrict__ W2b,
    _Float16* __restrict__ Wh, float* __restrict__ bias)
{
  __shared__ float w2row[C_DIM];
  __shared__ float wcb_s[C_DIM];
  const int d = blockIdx.x;
  const int c = threadIdx.x;
  w2row[c] = W2w[d * C_DIM + c];
  wcb_s[c] = Wcb[c];
  __syncthreads();
  for (int k = 0; k < SLOTS; ++k) {
    float acc = (k == 0) ? W1w[d * C_DIM + c] : 0.0f;
    for (int e = 0; e < C_DIM; ++e)
      acc += w2row[e] * Wcw[(size_t)(e * C_DIM + c) * SLOTS + k];
    Wh[(size_t)d * KTOT + k * C_DIM + c] = (_Float16)acc;
  }
  if (c == 0) {
    float b = W1b[d] + W2b[d];
    for (int e = 0; e < C_DIM; ++e) b += w2row[e] * wcb_s[e];
    bias[d] = b;
  }
}

// ---------------------------------------------------------------------------
// Kernel 2: one-time x f32 -> f16 (halves the L2-side gather traffic).
// ---------------------------------------------------------------------------
__global__ __launch_bounds__(256) void cvt_f32_to_f16(
    const float* __restrict__ x, _Float16* __restrict__ xh)
{
  const size_t i = ((size_t)blockIdx.x * 256 + threadIdx.x) * 4;
  const float4 v = *(const float4*)(x + i);
  v4h h;
  h.x = (_Float16)v.x; h.y = (_Float16)v.y;
  h.z = (_Float16)v.z; h.w = (_Float16)v.w;
  *(v4h*)(xh + i) = h;
}

// ---------------------------------------------------------------------------
// Kernel 3: fused gather-GEMM.
// Block = 256 threads = 8 waves; each wave owns TWO 16-node tiles (M=32) and
// all 128 output channels, so each LDS B-fragment feeds 2 WMMAs (halves LDS
// bandwidth per FLOP). Weight slot (128x128 f16 = 32 KB) staged into LDS
// with GLOBAL_LOAD_ASYNC_TO_LDS_B128 (ASYNCcnt), double-buffered so slot j+1
// streams in while slot j computes. sched_barrier(0) between the B-load
// batch and the WMMA batch forces grouped ds_loads + back-to-back WMMAs.
// ---------------------------------------------------------------------------
__global__ __launch_bounds__(256) void lfe_main(
    const _Float16* __restrict__ xh, const int* __restrict__ adj,
    const _Float16* __restrict__ Wh, const float* __restrict__ bias,
    float* __restrict__ out)
{
  __shared__ _Float16 lw[2][C_DIM * C_DIM];   // 2 x 32 KB double buffer

  const int lane = threadIdx.x & 31;
  const int wave = threadIdx.x >> 5;
  const int m    = lane & 15;              // row within tile / channel lane
  const int hi   = lane >> 4;              // half-wave select
  const size_t tile0 = ((size_t)blockIdx.x * 8 + wave) * 32;
  const size_t tile1 = tile0 + 16;
  const size_t node0 = tile0 + m;          // 20000 % 32 == 0: no batch straddle
  const size_t node1 = tile1 + m;
  const int bbase = (int)((node0 / N_DIM) * N_DIM);
  const int* arow0 = adj + node0 * K_NB;
  const int* arow1 = adj + node1 * K_NB;

  // Per-thread staging slice: 128 B = 8 x b128 (256 threads cover 32 KB).
  const int sd = threadIdx.x >> 1;
  const int sc = (threadIdx.x & 1) * 64;

  // Async global->LDS stage of weight slot j into buffer `buf`.
  // INST_OFFSET is added to BOTH the LDS and global addresses (ISA 10.7).
  // LDS address via ptrtoint of the real shared pointer (flat addr bits
  // [31:0] are the wave-relative LDS address; also escapes `lw`).
  auto stage_async = [&](int j, int buf) {
    const unsigned long long gp =
        (unsigned long long)(Wh + (size_t)sd * KTOT + j * C_DIM + sc);
    const unsigned loff =
        (unsigned)(unsigned long long)(&lw[buf][sd * C_DIM + sc]);
#pragma unroll
    for (int i = 0; i < 8; ++i) {
      asm volatile("global_load_async_to_lds_b128 %0, %1, off offset:%2"
                   :
                   : "v"(loff), "v"(gp), "n"(i * 16)
                   : "memory");
    }
  };
  auto wait_async = [&]() {
    asm volatile("s_wait_asynccnt 0x0" ::: "memory");
  };

  stage_async(0, 0);   // kick off first weight slot

  v8f acc0[8] = {};
  v8f acc1[8] = {};
  int s0 = (int)node0, s1 = (int)node1;   // gather rows for slot 0 = self

  wait_async();
  __syncthreads();

  for (int j = 0; j < SLOTS; ++j) {
    const int buf = j & 1;
    if (j + 1 < SLOTS) stage_async(j + 1, buf ^ 1);

    // Prefetch next slot's neighbor indices (consumed next iteration).
    int n0 = 0, n1 = 0;
    if (j + 1 < SLOTS) { n0 = bbase + arow0[j]; n1 = bbase + arow1[j]; }

    const _Float16* lwc = &lw[buf][0];
    const _Float16* fr0 = xh + (size_t)s0 * C_DIM + hi * 8;
    const _Float16* fr1 = xh + (size_t)s1 * C_DIM + hi * 8;

#pragma unroll
    for (int q = 0; q < 4; ++q) {
      const int cb = q * 32;
      // A fragments (16x32 f16): lane holds K +0..7 and +16..23; upper
      // half-wave shifted by +8 (folded into fr*).
      v8h a00 = *(const v8h*)(fr0 + cb);
      v8h a01 = *(const v8h*)(fr0 + cb + 16);
      v16h A0 = __builtin_shufflevector(a00, a01,
          0,1,2,3,4,5,6,7,8,9,10,11,12,13,14,15);
      v8h a10 = *(const v8h*)(fr1 + cb);
      v8h a11 = *(const v8h*)(fr1 + cb + 16);
      v16h A1 = __builtin_shufflevector(a10, a11,
          0,1,2,3,4,5,6,7,8,9,10,11,12,13,14,15);

      // B fragments (32x16 f16): lane n = channel, K +0..7 and +16..23, upper
      // half-wave +8. One batch of 16 ds_load_b128 feeds 16 WMMAs.
      v16h Bf[8];
#pragma unroll
      for (int t = 0; t < 8; ++t) {
        const _Float16* bp = lwc + (size_t)(t * 16 + m) * C_DIM + cb + hi * 8;
        v8h b0 = *(const v8h*)bp;
        v8h b1 = *(const v8h*)(bp + 16);
        Bf[t] = __builtin_shufflevector(b0, b1,
            0,1,2,3,4,5,6,7,8,9,10,11,12,13,14,15);
      }
      __builtin_amdgcn_sched_barrier(0);  // loads grouped before WMMA burst
#pragma unroll
      for (int t = 0; t < 8; ++t) {
        acc0[t] = __builtin_amdgcn_wmma_f32_16x16x32_f16(
            false, A0, false, Bf[t], (short)0, acc0[t], false, false);
        acc1[t] = __builtin_amdgcn_wmma_f32_16x16x32_f16(
            false, A1, false, Bf[t], (short)0, acc1[t], false, false);
      }
    }

    s0 = n0; s1 = n1;
    if (j + 1 < SLOTS) {
      wait_async();        // next slot's weights landed in LDS
      __syncthreads();     // all waves done reading current buffer
    }
  }

  // D layout: VGPR i -> row (tile + hi*8 + i), col = t*16 + m. Add bias, store.
#pragma unroll
  for (int t = 0; t < 8; ++t) {
    const int ch = t * 16 + m;
    const float bv = bias[ch];
#pragma unroll
    for (int i = 0; i < 8; ++i) {
      out[(tile0 + hi * 8 + i) * C_DIM + ch] = acc0[t][i] + bv;
      out[(tile1 + hi * 8 + i) * C_DIM + ch] = acc1[t][i] + bv;
    }
  }
}

// ---------------------------------------------------------------------------
extern "C" void kernel_launch(void* const* d_in, const int* in_sizes, int n_in,
                              void* d_out, int out_size, void* d_ws, size_t ws_size,
                              hipStream_t stream) {
  const float* x   = (const float*)d_in[0];
  const int*   adj = (const int*)d_in[1];
  const float* W1w = (const float*)d_in[2];
  const float* W1b = (const float*)d_in[3];
  const float* Wcw = (const float*)d_in[4];
  const float* Wcb = (const float*)d_in[5];
  const float* W2w = (const float*)d_in[6];
  const float* W2b = (const float*)d_in[7];
  float* out = (float*)d_out;

  char* ws = (char*)d_ws;
  const size_t xh_bytes = (size_t)B_DIM * N_DIM * C_DIM * sizeof(_Float16); // 40.96 MB
  const size_t wh_bytes = (size_t)C_DIM * KTOT * sizeof(_Float16);          // 352 KB
  _Float16* xh   = (_Float16*)ws;
  _Float16* Wh   = (_Float16*)(ws + xh_bytes);
  float*    bias = (float*)(ws + xh_bytes + wh_bytes);

  cvt_f32_to_f16<<<(B_DIM * N_DIM * C_DIM) / (256 * 4), 256, 0, stream>>>(x, xh);
  prep_weights<<<C_DIM, C_DIM, 0, stream>>>(W1w, W1b, Wcw, Wcb, W2w, W2b, Wh, bias);
  lfe_main<<<(B_DIM * N_DIM) / (8 * 32), 256, 0, stream>>>(xh, adj, Wh, bias, out);
}
